// L_GF_8211977470537
// MI455X (gfx1250) — matrix operand order repaired and checked by reading, not compiled, not run on hardware
//
#include <hip/hip_runtime.h>
#include <hip/hip_bf16.h>

// ---------------- problem constants ----------------
#define B_   2
#define L_   4096          // H*W
#define C_   128           // D_MODEL
#define D_   256           // D_INNER
#define NST  16            // D_STATE
#define RK   8             // DT_RANK
#define PROJW 40           // DT_RANK + 2*D_STATE
#define PROJP 48           // padded to 3 WMMA tiles
#define BL   (B_ * L_)     // 8192 tokens
#define PAD_H 8            // LDS row padding in halves (4 DWORDs) for bank-conflict avoidance

typedef __attribute__((ext_vector_type(16))) _Float16 v16h;
typedef __attribute__((ext_vector_type(8)))  float    v8f;
typedef __attribute__((ext_vector_type(4)))  unsigned int u32x4;
typedef __attribute__((ext_vector_type(4)))  int      i32x4;
typedef __attribute__((ext_vector_type(8)))  int      i32x8;

#if defined(__has_builtin)
#if __has_builtin(__builtin_amdgcn_tensor_load_to_lds)
#define HAVE_TDM 1
#else
#define HAVE_TDM 0
#endif
#else
#define HAVE_TDM 0
#endif

// Opaque escape: forces the compiler to treat LDS as genuinely written (the TDM builtin
// only carries the LDS address as an integer inside the descriptor, so without this the
// LDS array has "no writers" and its reads fold to undef, as seen in round-2 asm).
__device__ __forceinline__ void lds_escape_fence(void* p) {
  asm volatile("" : "+v"(p) : : "memory");
}

// ---------------- TDM: 2D f16 tile -> LDS (descriptor per CDNA5 ISA) ----
#if HAVE_TDM
__device__ __forceinline__ void tdm_load_2d_f16(unsigned lds_off, const _Float16* gsrc,
                                                unsigned tile_d0 /*elems per row*/,
                                                unsigned tile_d1 /*rows*/,
                                                unsigned row_stride /*elems*/,
                                                unsigned padIntervalCode,  // pads after 2^(c+1) DWORDs
                                                unsigned padAmountCode) {  // (a+1) DWORDs of pad
  const unsigned long long ga = (unsigned long long)(uintptr_t)gsrc;
  u32x4 g0;
  g0[0] = 1u;                                                // count=1, user descriptor
  g0[1] = lds_off;                                           // lds_addr (bytes)
  g0[2] = (unsigned)(ga & 0xFFFFFFFFu);                      // global_addr[31:0] (bits 95:64)
  g0[3] = (unsigned)((ga >> 32) & 0x01FFFFFFu) | (2u << 30); // addr[56:32] | type=2 (image)
  i32x8 g1;
  g1[0] = (int)((1u << 16) |                                 // data_size=1 (2 bytes)
                (1u << 20) |                                 // pad_enable
                (padIntervalCode << 22) |
                (padAmountCode << 25));
  g1[1] = (int)((tile_d0 & 0xFFFFu) << 16);                  // tensor_dim0[15:0] @ bits 63:48
  g1[2] = (int)(((tile_d0 >> 16) & 0xFFFFu) |                // tensor_dim0[31:16]
                ((tile_d1 & 0xFFFFu) << 16));                // tensor_dim1[15:0]
  g1[3] = (int)(((tile_d1 >> 16) & 0xFFFFu) |                // tensor_dim1[31:16]
                ((tile_d0 & 0xFFFFu) << 16));                // tile_dim0
  g1[4] = (int)(tile_d1 & 0xFFFFu);                          // tile_dim1 ; tile_dim2=0
  g1[5] = (int)row_stride;                                   // tensor_dim0_stride[31:0]
  g1[6] = 0;                                                 // stride[47:32] | dim1_stride[15:0]
  g1[7] = 0;
  const i32x4 z4 = {0, 0, 0, 0};
#if __clang_major__ >= 23
  const i32x8 z8 = {0, 0, 0, 0, 0, 0, 0, 0};
  __builtin_amdgcn_tensor_load_to_lds(g0, g1, z4, z4, z8, 0);
#else
  __builtin_amdgcn_tensor_load_to_lds(g0, g1, z4, z4, 0);
#endif
}
#endif

// Stage `rows` rows of K contiguous f16 (transposed weight, row stride = K in global) into
// LDS with padded row stride K+PAD_H halves. sW must be the only __shared__ object (offset 0).
// kLog2: log2(K); padIntervalCode: K/2 DWORDs per row -> code = log2(K/2)-1.
__device__ __forceinline__ void stage_weightT(_Float16* sW, const _Float16* gWT,
                                              int rows, int K, int kLog2) {
#if HAVE_TDM
  if (threadIdx.x < 32) {
    const unsigned padIntervalCode = (unsigned)(kLog2 - 2);  // K=128 -> 5 (64 dw), K=256 -> 6
    const unsigned padAmountCode   = 3u;                     // 4 DWORDs = 8 halves
    tdm_load_2d_f16(0u, gWT, (unsigned)K, (unsigned)rows, (unsigned)K,
                    padIntervalCode, padAmountCode);
    __builtin_amdgcn_s_wait_tensorcnt(0);
  }
#else
  const int KP = K + PAD_H;
  for (int i = threadIdx.x; i < rows * K; i += blockDim.x) {
    const int r = i >> kLog2;
    const int k = i & (K - 1);
    sW[(size_t)r * KP + k] = gWT[i];
  }
#endif
  __syncthreads();
  lds_escape_fence(sW);
}

// ---------------- WMMA fragment helpers (CDNA5 wave32 layouts) ----------------
// A 16x32 f16: lanes 0-15 row=lane, K = {0..7 in v0..3, 16..23 in v4..7};
//              lanes 16-31 row=lane-16, K = {8..15, 24..31}.
__device__ __forceinline__ v16h load_a_frag(const float* A, int lda, int row0, int k0, int lane) {
  const int hlf = (lane >> 4) & 1;
  const int row = row0 + (lane & 15);
  const float* p = A + (size_t)row * lda + k0 + 8 * hlf;
  v16h a;
#pragma unroll
  for (int e = 0; e < 8; ++e) a[e] = (_Float16)p[e];
#pragma unroll
  for (int e = 0; e < 8; ++e) a[8 + e] = (_Float16)p[16 + e];
  return a;
}

// B 32x16 f16 from LDS holding the TRANSPOSED weight (row = output col, KP-strided):
// lane n<16: col=n, K={k0..k0+7, k0+16..k0+23} ; lane n>=16: col=n-16, K={k0+8..k0+15, k0+24..k0+31}.
// two runs of 8 contiguous halves per lane -> 2x ds_load_b128.
__device__ __forceinline__ v16h load_b_frag_ldsT(const _Float16* sB, int KP, int k0,
                                                 int colLocal, int lane) {
  const int hlf = (lane >> 4) & 1;
  const int col = colLocal + (lane & 15);
  const _Float16* p = sB + (size_t)col * KP + k0 + 8 * hlf;
  v16h b;
#pragma unroll
  for (int e = 0; e < 8; ++e) { b[e] = p[e]; b[8 + e] = p[16 + e]; }
  return b;
}

// C/D 16x16 f32: VGPR r -> (M = r + 8*(lane>=16), N = lane&15).
__device__ __forceinline__ void store_c_frag(float* Cm, int ldc, int row0, int col0,
                                             int lane, v8f c) {
  const int hlf = (lane >> 4) & 1;
  const int col = col0 + (lane & 15);
#pragma unroll
  for (int r = 0; r < 8; ++r)
    Cm[(size_t)(row0 + r + 8 * hlf) * ldc + col] = c[r];
}

__device__ __forceinline__ float silu_f(float v) { return v / (1.0f + __expf(-v)); }

// ---------------- 0) weight pre-conversion to f16, TRANSPOSED (N x K) ----------------
__global__ void prep_weights_kernel(const float* __restrict__ Winx, const float* __restrict__ Winz,
                                    const float* __restrict__ Wxp,  const float* __restrict__ Wxpb,
                                    const float* __restrict__ Wout,
                                    _Float16* __restrict__ hWinxT, _Float16* __restrict__ hWinzT,
                                    _Float16* __restrict__ hWxpT,  _Float16* __restrict__ hWxpbT,
                                    _Float16* __restrict__ hWoutT) {
  const int i = blockIdx.x * blockDim.x + threadIdx.x;
  if (i < D_ * C_) {                       // W_in: 128x256 -> T: 256 rows x 128
    const int c = i >> 7, k = i & (C_ - 1);
    hWinxT[i] = (_Float16)Winx[k * D_ + c];
    hWinzT[i] = (_Float16)Winz[k * D_ + c];
  }
  if (i < C_ * D_) {                       // W_out: 256x128 -> T: 128 rows x 256
    const int c = i >> 8, k = i & (D_ - 1);
    hWoutT[i] = (_Float16)Wout[k * C_ + c];
  }
  if (i < PROJP * D_) {                    // W_x_proj: 256x40 -> T padded: 48 rows x 256
    const int c = i >> 8, k = i & (D_ - 1);
    const float vf = (c < PROJW) ? Wxp[k * PROJW + c]  : 0.0f;
    const float vb = (c < PROJW) ? Wxpb[k * PROJW + c] : 0.0f;
    hWxpT[i]  = (_Float16)vf;
    hWxpbT[i] = (_Float16)vb;
  }
}

// ---------------- 1) LayerNorm over C=128 (one wave per token) ----------------
__global__ void ln_kernel(const float* __restrict__ x1, const float* __restrict__ x2,
                          const float* __restrict__ g1, const float* __restrict__ b1,
                          const float* __restrict__ g2, const float* __restrict__ b2,
                          float* __restrict__ nout) {
  const int wid  = (blockIdx.x * blockDim.x + threadIdx.x) >> 5;  // row id: t*BL + b*L + l
  const int lane = threadIdx.x & 31;
  const int t    = wid / BL;
  const int rem  = wid - t * BL;              // b*L + l
  const int b    = rem >> 12;                 // /L_
  const int l    = rem & (L_ - 1);
  const float* src = t ? x2 : x1;
  const float* g   = t ? g2 : g1;
  const float* bb  = t ? b2 : b1;

  float v[4];
  float sum = 0.0f;
#pragma unroll
  for (int i = 0; i < 4; ++i) {
    const int c = lane + 32 * i;
    const float val = src[(size_t)b * C_ * L_ + (size_t)c * L_ + l];
    v[i] = val; sum += val;
  }
#pragma unroll
  for (int off = 16; off; off >>= 1) sum += __shfl_xor(sum, off, 32);
  const float mean = sum * (1.0f / C_);
  float sq = 0.0f;
#pragma unroll
  for (int i = 0; i < 4; ++i) { const float d = v[i] - mean; sq += d * d; }
#pragma unroll
  for (int off = 16; off; off >>= 1) sq += __shfl_xor(sq, off, 32);
  const float rstd = rsqrtf(sq * (1.0f / C_) + 1e-5f);

  float* dst = nout + ((size_t)t * BL + rem) * C_;
#pragma unroll
  for (int i = 0; i < 4; ++i) {
    const int c = lane + 32 * i;
    dst[c] = (v[i] - mean) * rstd * g[c] + bb[c];
  }
}

// ---------------- 2) input projections x = n@W_in_x, z = n@W_in_z (WMMA + TDM) ------------
// grid.z = 4: zi = f*2 + isZ ; M=8192, K=128, N=256; block = 4 waves = 64-col stripe
__global__ void inproj_kernel(const float* __restrict__ nin,
                              const _Float16* __restrict__ WxT, const _Float16* __restrict__ WzT,
                              float* __restrict__ xbuf, float* __restrict__ zbuf) {
  __shared__ _Float16 sW[64 * (C_ + PAD_H)];   // 64 cols x (128+8) halves = 17 KB

  const int zi = blockIdx.z;
  const int f = zi >> 1, isZ = zi & 1;
  const int t = isZ ^ f;                       // which normalized tensor feeds this product
  const float* A = nin + (size_t)t * BL * C_;
  const _Float16* WT = isZ ? WzT : WxT;        // transposed: 256 rows x 128
  float* Dst = (isZ ? zbuf : xbuf) + (size_t)f * BL * D_;

  const int colBase = blockIdx.y * 64;
  stage_weightT(sW, WT + (size_t)colBase * C_, 64, C_, 7);

  const int lane = threadIdx.x & 31;
  const int w    = threadIdx.x >> 5;
  const int row0 = blockIdx.x * 16;
  const int colL = w * 16;                     // local column tile within stripe

  v8f acc = {};
#pragma unroll
  for (int k0 = 0; k0 < C_; k0 += 32) {
    v16h af = load_a_frag(A, C_, row0, k0, lane);
    v16h bf = load_b_frag_ldsT(sW, C_ + PAD_H, k0, colL, lane);
    acc = __builtin_amdgcn_wmma_f32_16x16x32_f16(false, af, false, bf, (short)0, acc, false, false);
  }
  store_c_frag(Dst, D_, row0, colBase + colL, lane, acc);
}

// ---------------- 3) causal depthwise conv (fwd + reversed) + SiLU ----------------
// grid.z = 4: zi = f*2 + dir
__global__ void conv_silu_kernel(const float* __restrict__ xbuf,
                                 const float* __restrict__ cw_f, const float* __restrict__ cb_f,
                                 const float* __restrict__ cw_b, const float* __restrict__ cb_b,
                                 float* __restrict__ xc) {
  const int zi = blockIdx.z;
  const int f = zi >> 1, dir = zi & 1;
  const size_t idx = (size_t)blockIdx.x * blockDim.x + threadIdx.x;  // over BL*D
  const int d = (int)(idx & (D_ - 1));
  const size_t bl = idx >> 8;                  // b*L + l  (in branch coordinates)
  const int l = (int)(bl & (L_ - 1));
  const int b = (int)(bl >> 12);

  const float* cw = dir ? cw_b : cw_f;
  const float* cb = dir ? cb_b : cb_f;
  const float* xsrc = xbuf + (size_t)f * BL * D_ + (size_t)b * L_ * D_;

  float acc = cb[d];
#pragma unroll
  for (int j = 0; j < 4; ++j) {
    const int lj = l - 3 + j;                  // position in (possibly reversed) sequence
    if (lj >= 0) {
      const int lsrc = dir ? (L_ - 1 - lj) : lj;
      acc += cw[d * 4 + j] * xsrc[(size_t)lsrc * D_ + d];
    }
  }
  xc[((size_t)zi * BL + bl) * D_ + d] = silu_f(acc);
}

// ---------------- 4) x-projection dbl = xc @ W_x_proj (WMMA + TDM, N=48 padded) -----------
// grid.z = 4: zi = f*2 + dir ; block = 96 threads (3 waves -> 3 col tiles)
__global__ void xproj_kernel(const float* __restrict__ xc,
                             const _Float16* __restrict__ WfT, const _Float16* __restrict__ WbT,
                             float* __restrict__ dbl) {
  __shared__ _Float16 sW[PROJP * (D_ + PAD_H)];  // 48 x 264 halves = 25 KB

  const int zi = blockIdx.z;
  const int dir = zi & 1;
  const float* A = xc + (size_t)zi * BL * D_;
  const _Float16* WT = dir ? WbT : WfT;          // transposed padded: 48 rows x 256

  stage_weightT(sW, WT, PROJP, D_, 8);

  const int lane = threadIdx.x & 31;
  const int w    = threadIdx.x >> 5;             // 0..2
  const int row0 = blockIdx.x * 16;
  const int colL = w * 16;

  v8f acc = {};
#pragma unroll
  for (int k0 = 0; k0 < D_; k0 += 32) {
    v16h af = load_a_frag(A, D_, row0, k0, lane);
    v16h bf = load_b_frag_ldsT(sW, D_ + PAD_H, k0, colL, lane);
    acc = __builtin_amdgcn_wmma_f32_16x16x32_f16(false, af, false, bf, (short)0, acc, false, false);
  }
  store_c_frag(dbl + (size_t)zi * BL * PROJP, PROJP, row0, colL, lane, acc);
}

// ---------------- 5) selective scan (register-resident state, serial over L) --------------
// grid = (B, dir, f), block = 256 (thread = channel d). dt fused in (K=8 dot + softplus).
__global__ void scan_kernel(const float* __restrict__ xc, const float* __restrict__ dbl,
                            const float* __restrict__ Wdt_f, const float* __restrict__ bdt_f,
                            const float* __restrict__ Alog_f, const float* __restrict__ Dp_f,
                            const float* __restrict__ Wdt_b, const float* __restrict__ bdt_b,
                            const float* __restrict__ Alog_b, const float* __restrict__ Dp_b,
                            float* __restrict__ ybuf) {
  const int b = blockIdx.x, dir = blockIdx.y, f = blockIdx.z;
  const int zi = f * 2 + dir;
  const int d = threadIdx.x;

  const float* Wdt  = dir ? Wdt_b  : Wdt_f;    // (8,256)
  const float* bdt  = dir ? bdt_b  : bdt_f;
  const float* Alog = dir ? Alog_b : Alog_f;   // (256,16)
  const float* Dpp  = dir ? Dp_b   : Dp_f;

  float wdt[RK];
#pragma unroll
  for (int r = 0; r < RK; ++r) wdt[r] = Wdt[r * D_ + d];
  float An[NST];
#pragma unroll
  for (int n = 0; n < NST; ++n) An[n] = -__expf(Alog[d * NST + n]);
  const float bdtd = bdt[d], Dpd = Dpp[d];

  const float*  u0 = xc   + ((size_t)zi * B_ + b) * L_ * D_ + d;
  const float4* q0 = (const float4*)(dbl + ((size_t)zi * B_ + b) * L_ * PROJP);
  float*        y0 = ybuf + ((size_t)zi * B_ + b) * L_ * D_ + d;

  float h[NST];
#pragma unroll
  for (int n = 0; n < NST; ++n) h[n] = 0.0f;

  for (int l = 0; l < L_; ++l) {
    // prefetch the streaming per-channel input a few steps ahead (global_prefetch_b8)
    if (l + 8 < L_) __builtin_prefetch(u0 + (size_t)(l + 8) * D_, 0, 1);

    const float u = u0[(size_t)l * D_];
    const float4* q = q0 + (size_t)l * (PROJP / 4);   // 12 float4 per row

    // dt = softplus(dbl[:,0:8] @ Wdt[:,d] + b_dt[d])
    const float4 r01 = q[0], r23 = q[1];
    float dtp = bdtd;
    dtp += r01.x * wdt[0] + r01.y * wdt[1] + r01.z * wdt[2] + r01.w * wdt[3];
    dtp += r23.x * wdt[4] + r23.y * wdt[5] + r23.z * wdt[6] + r23.w * wdt[7];
    const float dt = (dtp > 20.0f) ? dtp : __logf(1.0f + __expf(dtp));
    const float du = dt * u;

    float Bv[NST], Cv[NST];
    {
      float4 tq;
      tq = q[2]; Bv[0] = tq.x; Bv[1] = tq.y; Bv[2]  = tq.z; Bv[3]  = tq.w;
      tq = q[3]; Bv[4] = tq.x; Bv[5] = tq.y; Bv[6]  = tq.z; Bv[7]  = tq.w;
      tq = q[4]; Bv[8] = tq.x; Bv[9] = tq.y; Bv[10] = tq.z; Bv[11] = tq.w;
      tq = q[5]; Bv[12]= tq.x; Bv[13]= tq.y; Bv[14] = tq.z; Bv[15] = tq.w;
      tq = q[6]; Cv[0] = tq.x; Cv[1] = tq.y; Cv[2]  = tq.z; Cv[3]  = tq.w;
      tq = q[7]; Cv[4] = tq.x; Cv[5] = tq.y; Cv[6]  = tq.z; Cv[7]  = tq.w;
      tq = q[8]; Cv[8] = tq.x; Cv[9] = tq.y; Cv[10] = tq.z; Cv[11] = tq.w;
      tq = q[9]; Cv[12]= tq.x; Cv[13]= tq.y; Cv[14] = tq.z; Cv[15] = tq.w;
    }

    float acc = 0.0f;
#pragma unroll
    for (int n = 0; n < NST; ++n) {
      h[n] = __expf(dt * An[n]) * h[n] + du * Bv[n];
      acc += h[n] * Cv[n];
    }
    const float yv = acc + u * Dpd;

    const int lout = dir ? (L_ - 1 - l) : l;   // backward branch un-reverses on store
    y0[(size_t)lout * D_] = yv;
  }
}

// ---------------- 6) gate + out-projection + transpose-store + skip (WMMA + TDM) ----------
// A_eff[m][k] = (yf+yb) * silu(z), computed on the fly. M=8192, K=256, N=128.
__global__ void outproj_kernel(const float* __restrict__ ybuf, const float* __restrict__ zbuf,
                               const _Float16* __restrict__ WoutT,
                               const float* __restrict__ x1, const float* __restrict__ x2,
                               const float* __restrict__ skip, float* __restrict__ out) {
  __shared__ _Float16 sW[64 * (D_ + PAD_H)];   // 64 cols x 264 halves = 33 KB

  const int f = blockIdx.z;
  const int colBase = blockIdx.y * 64;
  stage_weightT(sW, WoutT + (size_t)colBase * D_, 64, D_, 8);

  const int lane = threadIdx.x & 31;
  const int w    = threadIdx.x >> 5;
  const int row0 = blockIdx.x * 16;
  const int colL = w * 16;

  const float* yf = ybuf + (size_t)(f * 2 + 0) * BL * D_;
  const float* yb = ybuf + (size_t)(f * 2 + 1) * BL * D_;
  const float* zz = zbuf + (size_t)f * BL * D_;

  const int hlf = (lane >> 4) & 1;
  const int arow = row0 + (lane & 15);

  v8f acc = {};
#pragma unroll
  for (int k0 = 0; k0 < D_; k0 += 32) {
    const size_t base = (size_t)arow * D_ + k0 + 8 * hlf;
    v16h af;
#pragma unroll
    for (int e = 0; e < 8; ++e) {
      { const size_t i = base + e;
        const float zv = zz[i];
        af[e]     = (_Float16)((yf[i] + yb[i]) * silu_f(zv)); }
      { const size_t i = base + 16 + e;
        const float zv = zz[i];
        af[8 + e] = (_Float16)((yf[i] + yb[i]) * silu_f(zv)); }
    }
    v16h bf = load_b_frag_ldsT(sW, D_ + PAD_H, k0, colL, lane);
    acc = __builtin_amdgcn_wmma_f32_16x16x32_f16(false, af, false, bf, (short)0, acc, false, false);
  }

  // D tile element (m, n): token m -> (b, l); store transposed to out[f][b][n][l] + skip
  const float* xin = f ? x2 : x1;
  const float ss = skip[0];
  const int n = colBase + colL + (lane & 15);
#pragma unroll
  for (int r = 0; r < 8; ++r) {
    const int m = row0 + r + 8 * hlf;
    const int b = m >> 12;                     // /L_
    const int l = m & (L_ - 1);
    const size_t oi = (size_t)b * C_ * L_ + (size_t)n * L_ + l;
    out[(size_t)f * B_ * C_ * L_ + oi] = acc[r] + ss * xin[oi];
  }
}

// ---------------- host launcher ----------------
extern "C" void kernel_launch(void* const* d_in, const int* in_sizes, int n_in,
                              void* d_out, int out_size, void* d_ws, size_t ws_size,
                              hipStream_t stream) {
  const float* x1    = (const float*)d_in[0];
  const float* x2    = (const float*)d_in[1];
  const float* ln1g  = (const float*)d_in[2];
  const float* ln1b  = (const float*)d_in[3];
  const float* ln2g  = (const float*)d_in[4];
  const float* ln2b  = (const float*)d_in[5];
  const float* skip  = (const float*)d_in[6];
  const float* Winx  = (const float*)d_in[7];
  const float* Winz  = (const float*)d_in[8];
  const float* cw    = (const float*)d_in[9];
  const float* cb    = (const float*)d_in[10];
  const float* Wxp   = (const float*)d_in[11];
  const float* Wdt   = (const float*)d_in[12];
  const float* bdt   = (const float*)d_in[13];
  const float* Alog  = (const float*)d_in[14];
  const float* Dp    = (const float*)d_in[15];
  const float* cwb   = (const float*)d_in[16];
  const float* cbb   = (const float*)d_in[17];
  const float* Wxpb  = (const float*)d_in[18];
  const float* Wdtb  = (const float*)d_in[19];
  const float* bdtb  = (const float*)d_in[20];
  const float* Alogb = (const float*)d_in[21];
  const float* Dpb   = (const float*)d_in[22];
  const float* Wout  = (const float*)d_in[23];

  // workspace layout (floats, then transposed f16 weight buffers)
  float* ws   = (float*)d_ws;
  float* nrm  = ws;                                 // 2 * BL * C_
  float* xb   = nrm  + (size_t)2 * BL * C_;         // 2 * BL * D_
  float* zb   = xb   + (size_t)2 * BL * D_;         // 2 * BL * D_
  float* xc   = zb   + (size_t)2 * BL * D_;         // 4 * BL * D_
  float* dbl  = xc   + (size_t)4 * BL * D_;         // 4 * BL * PROJP
  float* ybuf = dbl  + (size_t)4 * BL * PROJP;      // 4 * BL * D_
  _Float16* hbase  = (_Float16*)(ybuf + (size_t)4 * BL * D_);
  _Float16* hWinxT = hbase;                         // 256 x 128
  _Float16* hWinzT = hWinxT + D_ * C_;              // 256 x 128
  _Float16* hWoutT = hWinzT + D_ * C_;              // 128 x 256
  _Float16* hWxpT  = hWoutT + C_ * D_;              // 48 x 256
  _Float16* hWxpbT = hWxpT  + PROJP * D_;           // 48 x 256

  float* out = (float*)d_out;

  // 0) weight conversion to f16 + transpose (+ padding for x-proj)
  prep_weights_kernel<<<dim3((C_ * D_) / 256), 256, 0, stream>>>(
      Winx, Winz, Wxp, Wxpb, Wout, hWinxT, hWinzT, hWxpT, hWxpbT, hWoutT);

  // 1) LayerNorm: one wave per token, 8 waves per block
  ln_kernel<<<dim3(2 * BL / 8), 256, 0, stream>>>(x1, x2, ln1g, ln1b, ln2g, ln2b, nrm);

  // 2) input projections (4 GEMMs, M=8192 K=128 N=256)
  inproj_kernel<<<dim3(BL / 16, D_ / 64, 4), 128, 0, stream>>>(nrm, hWinxT, hWinzT, xb, zb);

  // 3) causal depthwise conv + SiLU (fwd + reversed branches)
  conv_silu_kernel<<<dim3((BL * D_) / 256, 1, 4), 256, 0, stream>>>(xb, cw, cb, cwb, cbb, xc);

  // 4) x-projection (M=8192 K=256 N=48 padded)
  xproj_kernel<<<dim3(BL / 16, 1, 4), 96, 0, stream>>>(xc, hWxpT, hWxpbT, dbl);

  // 5) selective scan: 8 workgroups (b, dir, f), 256 channels each
  scan_kernel<<<dim3(B_, 2, 2), 256, 0, stream>>>(xc, dbl, Wdt, bdt, Alog, Dp,
                                                  Wdtb, bdtb, Alogb, Dpb, ybuf);

  // 6) gate + out-projection + transpose + skip (M=8192 K=256 N=128)
  outproj_kernel<<<dim3(BL / 16, C_ / 64, 2), 128, 0, stream>>>(ybuf, zb, hWoutT, x1, x2, skip, out);
}
